// MinCostMatcher_79250736545929
// MI455X (gfx1250) — matrix operand order repaired and checked by reading, not compiled, not run on hardware
//
#include <hip/hip_runtime.h>

// ---------------------------------------------------------------------------
// MinCostMatcher for MI455X (gfx1250, wave32).
//  - cls einsum 'bnc,bmc->bmn' on v_wmma_f32_16x16x32_bf16 (A = ct is exactly
//    {0,1} in bf16, f32 accumulate; only diff is rounded to bf16).
//  - ct is pre-swizzled to the WMMA A-fragment layout once per batch, then
//    DMA'd into LDS with the Tensor Data Mover (tensor_load_to_lds +
//    s_wait_tensorcnt), issued by wave 0 only.
//  - focal cost uses native v_log_f32 (log2) with ln2 folded into the algebra.
//  - L1 + GIoU + argmin-over-n fused in registers; (8,100,16384) cost tensor
//    never touches memory. Min reduction: order-preserving (f32,n) u64 keys,
//    shfl_xor within 16-lane half, ds atomic_min, then global atomic_min_u64.
//  - __launch_bounds__(128, 4) caps VGPRs at 256: no s_set_vgpr_msb traffic,
//    >=4 waves/SIMD occupancy.
// ---------------------------------------------------------------------------

typedef __attribute__((ext_vector_type(16))) __bf16       v16bf;
typedef __attribute__((ext_vector_type(8)))  float        v8f;
typedef __attribute__((ext_vector_type(4)))  unsigned int u32x4;
typedef __attribute__((ext_vector_type(8)))  int          i32x8;
typedef __attribute__((ext_vector_type(4)))  int          i32x4;

static constexpr int B_ = 8;       // batches
static constexpr int N_ = 16384;   // predictions
static constexpr int C_ = 80;      // classes
static constexpr int M_ = 100;     // gt boxes
static constexpr int MT = 7;       // m tiles of 16 -> 112 padded rows
static constexpr int KS = 3;       // k steps of 32 -> 96 padded classes
static constexpr int NITER   = 2;  // n-subtiles (of 64) per workgroup
static constexpr int WG_N    = 64; // 4 waves x 16 columns
static constexpr int NGROUPS = N_ / (WG_N * NITER);      // 128
static constexpr int A_HALVES = MT * KS * 32 * 16;       // 10752 bf16 / batch
static constexpr int A_BYTES  = A_HALVES * 2;            // 21504 B
static constexpr int A_DWORDS = A_BYTES / 4;             // 5376
static constexpr size_t WS_A_OFF = 8192;                 // slots live in [0,6400)

union BfVec { v16bf v; unsigned short u[16]; };

__device__ __forceinline__ unsigned short f2bf_rne(float f) {
  unsigned u = __float_as_uint(f);
  unsigned r = (u + 0x7FFFu + ((u >> 16) & 1u)) >> 16;
  return (unsigned short)r;
}

__global__ void mcm_init_slots(unsigned long long* slots) {
  int i = blockIdx.x * blockDim.x + threadIdx.x;
  if (i < B_ * M_) slots[i] = 0xFFFFFFFFFFFFFFFFULL;
}

// Swizzle ct = (cls_true==1) into the WMMA 16-bit A-fragment layout, bf16,
// frag-major [t][ks][lane][16 halves], one 21504B block per batch.
// A 16x32 bf16 order used here: lane = M%16 + 16*(K>>4), half = K&15 (matches B).
__global__ __launch_bounds__(256)
void mcm_prep(const int* __restrict__ cls_true,
              unsigned short* __restrict__ wsA) {
  const int b = blockIdx.x;
  const int* ctb = cls_true + (size_t)b * M_ * C_;
  for (int idx = threadIdx.x; idx < A_HALVES; idx += 256) {
    int s   = idx & 15;
    int ln  = (idx >> 4) & 31;
    int tks = idx >> 9;
    int ks  = tks % KS;
    int t   = tks / KS;
    int m   = t * 16 + (ln & 15);
    int hi  = ln >> 4;
    int k   = ks * 32 + 16 * hi + s;
    unsigned short v = 0;
    if (m < M_ && k < C_ && ctb[m * C_ + k] == 1) v = 0x3F80;  // bf16(1.0)
    wsA[(size_t)b * A_HALVES + idx] = v;
  }
}

__global__ __launch_bounds__(128, 4)
void mcm_main(const float* __restrict__ cls_pred,
              const float* __restrict__ loc_pred,
              const float* __restrict__ loc_true,
              const unsigned short* __restrict__ wsA,
              unsigned long long* __restrict__ gslots) {
  __shared__ BfVec sA[MT * KS * 32];            // 21504 B, TDM destination
  __shared__ float sLT[112 * 4];                // loc_true staging
  __shared__ unsigned long long sSlot[112];     // per-m running min key

  const int b    = blockIdx.y;
  const int tid  = threadIdx.x;
  const int lane = tid & 31;
  const int w    = tid >> 5;
  const int grpBase = blockIdx.x * (WG_N * NITER);

  // ---- TDM: DMA the pre-swizzled A block for this batch into LDS --------
  // Wave 0 only: waves 1-3 skip via execz branch (never fetch the op).
  if (tid < 32) {
    // generic->LDS: low 32 bits of a generic LDS address are the LDS offset
    unsigned ldsOff = (unsigned)(unsigned long long)(void*)&sA[0];
    unsigned long long gaddr =
        (unsigned long long)(const void*)(wsA + (size_t)b * A_HALVES);
    // D# group0: count=1 | lds_addr | global_addr[56:0] | type=2
    u32x4 g0 = {1u, ldsOff, (unsigned)gaddr,
                ((unsigned)(gaddr >> 32) & 0x01FFFFFFu) | 0x80000000u};
    // D# group1: data_size=4B; tensor_dim0=tile_dim0=5376 dwords; dims1=1
    i32x8 g1 = {(int)0x00020000,            // data_size=2 (4B)
                (int)(A_DWORDS << 16),      // tensor_dim0[15:0] @ bits63:48
                (int)(1u << 16),            // dim0 hi=0, tensor_dim1=1
                (int)(A_DWORDS << 16),      // dim1 hi=0, tile_dim0=5376
                1,                          // tile_dim1=1, tile_dim2=0
                A_DWORDS,                   // tensor_dim0_stride
                0, 0};
    i32x4 gz4 = {0, 0, 0, 0};
    i32x8 gz8 = {0, 0, 0, 0, 0, 0, 0, 0};
    __builtin_amdgcn_tensor_load_to_lds(g0, g1, gz4, gz4, gz8, 0);
  }

  for (int idx = tid; idx < 112 * 4; idx += 128) {
    int m = idx >> 2;
    sLT[idx] = (m < M_) ? loc_true[((size_t)b * M_ + m) * 4 + (idx & 3)] : 0.f;
  }
  for (int idx = tid; idx < 112; idx += 128) sSlot[idx] = 0xFFFFFFFFFFFFFFFFULL;

  __builtin_amdgcn_s_wait_tensorcnt(0);
  __syncthreads();

  const int nCol0 = lane & 15;   // N within 16-wide subtile
  const int chalf = lane >> 4;   // which K-half this lane supplies for B

  #pragma unroll 1
  for (int it = 0; it < NITER; ++it) {
    const int n = grpBase + it * WG_N + w * 16 + nCol0;  // this lane's column

    if (it + 1 < NITER) {  // prefetch next subtile's cls_pred rows
      __builtin_prefetch(cls_pred + ((size_t)b * N_ + n + WG_N) * C_, 0, 1);
    }

    v8f acc[MT] = {};  // f32 accumulators, 7 m-tiles x (16x16)

    #pragma unroll 1
    for (int ks = 0; ks < KS; ++ks) {
      // ---- B fragment: diff[n, k] bf16, B 32x16 layout -----------------
      // lane = N%16 + 16*(K/16), half = K%16; a lane's 16 classes are
      // either all real (cbase<80) or all zero padding.
      BfVec bv;
      const int cbase = ks * 32 + chalf * 16;
      if (cbase < C_) {
        const float4* pv = (const float4*)
            (cls_pred + ((size_t)b * N_ + n) * C_ + cbase);
        float4 q[4] = {pv[0], pv[1], pv[2], pv[3]};
        const float* qf = (const float*)q;
        #pragma unroll
        for (int j = 0; j < 16; ++j) {
          float p   = qf[j];
          float omp = 1.f - p;
          // pos-neg focal cost, ln2 folded in; v_log_f32 = log2
          float l2p = __builtin_amdgcn_logf(p   + 1e-8f);
          float l2o = __builtin_amdgcn_logf(omp + 1e-8f);
          float d = 0.69314718056f * (0.75f * p * p * l2o
                                      - 0.25f * omp * omp * l2p);
          bv.u[j] = f2bf_rne(d);
        }
      } else {
        #pragma unroll
        for (int j = 0; j < 16; ++j) bv.u[j] = 0;
      }
      // ---- 7 WMMAs: D[m,n] += ct[m,k] * diff[n,k] ----------------------
      #pragma unroll
      for (int t = 0; t < MT; ++t) {
        v16bf a = sA[(t * KS + ks) * 32 + lane].v;
        acc[t] = __builtin_amdgcn_wmma_f32_16x16x32_bf16(
            false, a, false, bv.v, (short)0, acc[t], false, false);
      }
    }

    // ---- fused epilogue: L1 + GIoU + argmin reduction ------------------
    const float* lpp = loc_pred + ((size_t)b * N_ + n) * 4;
    float py0 = lpp[0], px0 = lpp[1], py1 = lpp[2], px1 = lpp[3];
    float parea = fmaxf(py1 - py0, 0.f) * fmaxf(px1 - px0, 0.f);

    #pragma unroll
    for (int t = 0; t < MT; ++t) {
      #pragma unroll
      for (int r = 0; r < 8; ++r) {
        int m = t * 16 + r + 8 * chalf;   // C/D layout: hi lanes hold M=r+8
        float ty0 = sLT[m * 4 + 0], tx0 = sLT[m * 4 + 1];
        float ty1 = sLT[m * 4 + 2], tx1 = sLT[m * 4 + 3];

        float reg = fabsf(ty0 - py0) + fabsf(tx0 - px0) +
                    fabsf(ty1 - py1) + fabsf(tx1 - px1);

        float tarea = fmaxf(ty1 - ty0, 0.f) * fmaxf(tx1 - tx0, 0.f);
        float iy0 = fmaxf(py0, ty0), ix0 = fmaxf(px0, tx0);
        float iy1 = fminf(py1, ty1), ix1 = fminf(px1, tx1);
        float inter = fmaxf(iy1 - iy0, 0.f) * fmaxf(ix1 - ix0, 0.f);
        float uni = parea + tarea - inter;
        float iou = (uni > 0.f) ? inter / uni : 0.f;
        float ey0 = fminf(py0, ty0), ex0 = fminf(px0, tx0);
        float ey1 = fmaxf(py1, ty1), ex1 = fmaxf(px1, tx1);
        float enc = fmaxf(ey1 - ey0, 0.f) * fmaxf(ex1 - ex0, 0.f);
        float giou = iou - ((enc > 0.f) ? (enc - uni) / enc : 0.f);

        float total = 2.f * acc[t][r] + 5.f * reg + 2.f * (1.f - giou);

        // order-preserving float key | n (ties -> smaller n = first index)
        unsigned fu = __float_as_uint(total);
        fu = (fu & 0x80000000u) ? ~fu : (fu | 0x80000000u);
        unsigned long long key =
            ((unsigned long long)fu << 32) | (unsigned)n;

        // min over the 16 lanes sharing this m (stays within half-wave)
        #pragma unroll
        for (int msk = 1; msk <= 8; msk <<= 1) {
          unsigned lo  = __shfl_xor((unsigned)key, msk, 32);
          unsigned hi2 = __shfl_xor((unsigned)(key >> 32), msk, 32);
          unsigned long long o = ((unsigned long long)hi2 << 32) | lo;
          key = (o < key) ? o : key;
        }
        if (nCol0 == 0 && m < M_) atomicMin(&sSlot[m], key);
      }
    }
  }

  __syncthreads();
  for (int m = tid; m < M_; m += 128)
    atomicMin(&gslots[b * M_ + m], sSlot[m]);
}

__global__ void mcm_finalize(const unsigned long long* __restrict__ gslots,
                             const int* __restrict__ cls_true,
                             int* __restrict__ out) {
  int i = blockIdx.x * blockDim.x + threadIdx.x;
  if (i >= B_ * M_) return;
  int b = i / M_, m = i % M_;
  int n = (int)(gslots[i] & 0xFFFFFFFFULL);
  const int* ct = cls_true + ((size_t)b * M_ + m) * C_;
  int cid = 0;
  for (int c = 0; c < C_; ++c) {
    if (ct[c] == 1) { cid = c; break; }   // argmax of {0,1}: first 1, else 0
  }
  out[i * 3 + 0] = b;
  out[i * 3 + 1] = n;
  out[i * 3 + 2] = cid;
}

extern "C" void kernel_launch(void* const* d_in, const int* in_sizes, int n_in,
                              void* d_out, int out_size, void* d_ws, size_t ws_size,
                              hipStream_t stream) {
  const float* cls_pred = (const float*)d_in[0];
  const float* loc_pred = (const float*)d_in[1];
  const int*   cls_true = (const int*)d_in[2];
  const float* loc_true = (const float*)d_in[3];
  // d_in[4] = reg_mask: unused by the reference output.
  (void)in_sizes; (void)n_in; (void)out_size; (void)ws_size;

  unsigned long long* gslots = (unsigned long long*)d_ws;          // 800*8 B
  unsigned short* wsA = (unsigned short*)((char*)d_ws + WS_A_OFF); // 8*21504 B

  mcm_init_slots<<<(B_ * M_ + 255) / 256, 256, 0, stream>>>(gslots);
  mcm_prep<<<B_, 256, 0, stream>>>(cls_true, wsA);

  dim3 grid(NGROUPS, B_);
  mcm_main<<<grid, 128, 0, stream>>>(cls_pred, loc_pred, loc_true, wsA,
                                     gslots);

  mcm_finalize<<<(B_ * M_ + 255) / 256, 256, 0, stream>>>(gslots, cls_true,
                                                          (int*)d_out);
}
